// SparseConvBlock_87497073754162
// MI455X (gfx1250) — compile-verified, matrix operand order checked by reading, not run on hardware
//
#include <hip/hip_runtime.h>

typedef __attribute__((ext_vector_type(16))) _Float16 v16h;
typedef __attribute__((ext_vector_type(8)))  _Float16 v8h;
typedef __attribute__((ext_vector_type(8)))  float    v8f;

#define L_LEN   8192
#define C_IN    256
#define C_OUT   256
#define BATCH   16
#define BN_EPS  1e-5f
#define IPAD    264          // padded LDS stride in halves (bank-conflict free)

// ---------------- reductions (wave32) ----------------
__device__ __forceinline__ float waveSum(float v) {
#pragma unroll
  for (int off = 16; off > 0; off >>= 1) v += __shfl_xor(v, off, 32);
  return v;
}
__device__ __forceinline__ float waveMax(float v) {
#pragma unroll
  for (int off = 16; off > 0; off >>= 1) v = fmaxf(v, __shfl_xor(v, off, 32));
  return v;
}
__device__ __forceinline__ float blockSum(float v, float* sm) {
  int lane = threadIdx.x & 31, wid = threadIdx.x >> 5;
  v = waveSum(v);
  if (lane == 0) sm[wid] = v;
  __syncthreads();
  if (wid == 0) {
    float t = (lane < 8) ? sm[lane] : 0.0f;
    t = waveSum(t);
    if (lane == 0) sm[0] = t;
  }
  __syncthreads();
  float r = sm[0];
  __syncthreads();
  return r;
}
__device__ __forceinline__ float blockMax(float v, float* sm) {
  int lane = threadIdx.x & 31, wid = threadIdx.x >> 5;
  v = waveMax(v);
  if (lane == 0) sm[wid] = v;
  __syncthreads();
  if (wid == 0) {
    float t = (lane < 8) ? sm[lane] : -3.4e38f;
    t = waveMax(t);
    if (lane == 0) sm[0] = t;
  }
  __syncthreads();
  float r = sm[0];
  __syncthreads();
  return r;
}

// ---------------- 1) BatchNorm batch statistics ----------------
__global__ __launch_bounds__(256) void bn_stats_kernel(
    const float* __restrict__ x, const float* __restrict__ gamma,
    const float* __restrict__ beta, float* __restrict__ bn_scale,
    float* __restrict__ bn_shift) {
  __shared__ float sm[8];
  int c = blockIdx.x;
  float s = 0.0f, s2 = 0.0f;
  for (int idx = threadIdx.x; idx < BATCH * L_LEN; idx += 256) {
    int b = idx >> 13, l = idx & (L_LEN - 1);
    float v = x[(((size_t)(b * C_IN + c)) << 13) + l];
    s += v; s2 += v * v;
  }
  s  = blockSum(s, sm);
  s2 = blockSum(s2, sm);
  if (threadIdx.x == 0) {
    const float invN = 1.0f / (float)(BATCH * L_LEN);
    float mean = s * invN;
    float var  = s2 * invN - mean * mean;
    float inv  = rsqrtf(var + BN_EPS);
    float g = gamma[c];
    bn_scale[c] = g * inv;
    bn_shift[c] = beta[c] - mean * g * inv;
  }
}

// ---------------- 2) weight-norm scalar g/||v|| ----------------
__global__ __launch_bounds__(256) void wnorm_kernel(
    const float* __restrict__ v, const float* __restrict__ g,
    float* __restrict__ wscale) {
  __shared__ float sm[8];
  float s = 0.0f;
  for (int idx = threadIdx.x; idx < C_OUT * C_IN * 3; idx += 256) {
    float w = v[idx];
    s += w * w;
  }
  s = blockSum(s, sm);
  if (threadIdx.x == 0) wscale[0] = g[0] / sqrtf(s);
}

// ---------------- 3) pack scaled weights: wA[k][o][i] as f16 ----------------
__global__ __launch_bounds__(256) void pack_w_kernel(
    const float* __restrict__ v, const float* __restrict__ wscale,
    _Float16* __restrict__ wh) {
  int idx = blockIdx.x * 256 + threadIdx.x;        // [k][o][i], i fastest
  int k = idx >> 16;
  int o = (idx >> 8) & 255;
  int i = idx & 255;
  wh[idx] = (_Float16)(wscale[0] * v[(o * C_IN + i) * 3 + k]);
}

// ---------------- 4) fused BN + sparsemax (bisection, no sort) ----------------
// Output stored directly as f16: the conv converts to f16 for WMMA anyway,
// so this halves s traffic (64 MiB total -> fully L2-resident) at zero
// additional precision cost.
__global__ __launch_bounds__(256) void sparsemax_kernel(
    const float* __restrict__ x, const float* __restrict__ bn_scale,
    const float* __restrict__ bn_shift, _Float16* __restrict__ s) {
  __shared__ float sm[8];
  int row = blockIdx.x;                 // b*256 + c
  int c = row & (C_IN - 1);
  const float* xr = x + ((size_t)row << 13);
  _Float16* sr = s + ((size_t)row << 13);
  float sc = bn_scale[c], sh = bn_shift[c];
  int t = threadIdx.x;

  float z[32];
  float mx = -3.4e38f;
#pragma unroll
  for (int r = 0; r < 32; ++r) {
    z[r] = xr[t + 256 * r] * sc + sh;
    mx = fmaxf(mx, z[r]);
  }
  mx = blockMax(mx, sm);

  // f(tau)=sum(max(z-tau,0)) is monotone decreasing, piecewise linear;
  // root in [mx-1, mx]. 32 bisections -> interval 2^-32, below fp32 ulp.
  float lo = mx - 1.0f, hi = mx;
  for (int it = 0; it < 32; ++it) {
    float tau = 0.5f * (lo + hi);
    float acc = 0.0f;
#pragma unroll
    for (int r = 0; r < 32; ++r) acc += fmaxf(z[r] - tau, 0.0f);
    acc = blockSum(acc, sm);
    if (acc >= 1.0f) lo = tau; else hi = tau;
  }
  float tau = 0.5f * (lo + hi);
#pragma unroll
  for (int r = 0; r < 32; ++r)
    sr[t + 256 * r] = (_Float16)fmaxf(z[r] - tau, 0.0f);
}

// ---------------- 5) implicit-GEMM conv via WMMA f16 -> f32 ----------------
// block: 8 waves, tile 128(O) x 64(L); grid (L/64, 2, B)
__global__ __launch_bounds__(256) void conv_wmma_kernel(
    const _Float16* __restrict__ wh, const _Float16* __restrict__ s,
    float* __restrict__ y) {
  __shared__ _Float16 Bl[66 * IPAD] __attribute__((aligned(16)));

  int b = blockIdx.z;
  int obase = blockIdx.y * 128;
  int lbase = blockIdx.x * 64;
  const _Float16* sb = s + ((size_t)b << 21);       // b*256*8192

  // stage s tile (66 L cols incl. halo x 256 channels) into LDS,
  // layout Bl[lcol][i] with padded stride: b128 reads are conflict-free.
  for (int idx = threadIdx.x; idx < 66 * 256; idx += 256) {
    int j = idx % 66;                 // consecutive lanes -> consecutive l (coalesced)
    int i = idx / 66;
    int le = lbase - 1 + j;
    _Float16 val = (le >= 0 && le < L_LEN) ? sb[((size_t)i << 13) + le]
                                           : (_Float16)0.0f;
    Bl[j * IPAD + i] = val;
  }
  __syncthreads();

  int lane = threadIdx.x & 31;
  int wid  = threadIdx.x >> 5;
  int col  = lane & 15;
  int sel8 = (lane < 16) ? 0 : 8;     // 16-bit A/B fragment: lane-group K offset
  int o    = obase + wid * 16 + col;  // A-matrix row (M) for this lane

  v8f acc[4];
#pragma unroll
  for (int lt = 0; lt < 4; ++lt) acc[lt] = (v8f){};

  for (int k = 0; k < 3; ++k) {
    const _Float16* wb = wh + (k << 16) + o * 256;
    for (int ic = 0; ic < 256; ic += 32) {
      // A fragment: K = ic + {0..7,16..23} (lanes 0-15) / {8..15,24..31} (lanes 16-31)
      v8h al = *(const v8h*)(wb + ic + sel8);
      v8h ah = *(const v8h*)(wb + ic + 16 + sel8);
      v16h a = __builtin_shufflevector(al, ah, 0,1,2,3,4,5,6,7,8,9,10,11,12,13,14,15);
#pragma unroll
      for (int lt = 0; lt < 4; ++lt) {
        int j = lt * 16 + col + k;    // l_eff = lbase + l_local + k - 1
        const _Float16* bp = &Bl[j * IPAD + ic + sel8];
        v8h b0 = *(const v8h*)bp;
        v8h b1 = *(const v8h*)(bp + 16);
        v16h bm = __builtin_shufflevector(b0, b1, 0,1,2,3,4,5,6,7,8,9,10,11,12,13,14,15);
        acc[lt] = __builtin_amdgcn_wmma_f32_16x16x32_f16(
            false, a, false, bm, (short)0, acc[lt], false, false);
      }
    }
  }

  // epilogue: ReLU + store. C/D layout: lanes 0-15 -> M=r, lanes 16-31 -> M=8+r
  int mbase = (lane < 16) ? 0 : 8;
#pragma unroll
  for (int lt = 0; lt < 4; ++lt) {
    int ll = lbase + lt * 16 + col;
#pragma unroll
    for (int r = 0; r < 8; ++r) {
      int oo = obase + wid * 16 + mbase + r;
      y[(((size_t)(b * C_OUT + oo)) << 13) + ll] = fmaxf(acc[lt][r], 0.0f);
    }
  }
}

// ---------------- host ----------------
extern "C" void kernel_launch(void* const* d_in, const int* in_sizes, int n_in,
                              void* d_out, int out_size, void* d_ws, size_t ws_size,
                              hipStream_t stream) {
  const float* x     = (const float*)d_in[0];   // [16,256,8192]
  const float* gamma = (const float*)d_in[1];   // [256]
  const float* beta  = (const float*)d_in[2];   // [256]
  const float* v     = (const float*)d_in[3];   // [256,256,3]
  const float* g     = (const float*)d_in[4];   // scalar

  char* ws = (char*)d_ws;
  float*    bn_scale = (float*)(ws + 0);        // 256 f32
  float*    bn_shift = (float*)(ws + 1024);     // 256 f32
  float*    wscale   = (float*)(ws + 2048);     // 1 f32
  _Float16* wh       = (_Float16*)(ws + 4096);  // 3*256*256 f16 = 384 KiB
  _Float16* s        = (_Float16*)(ws + 524288);// 16*256*8192 f16 = 64 MiB

  bn_stats_kernel<<<C_IN, 256, 0, stream>>>(x, gamma, beta, bn_scale, bn_shift);
  wnorm_kernel<<<1, 256, 0, stream>>>(v, g, wscale);
  pack_w_kernel<<<(3 * C_OUT * C_IN) / 256, 256, 0, stream>>>(v, wscale, wh);
  sparsemax_kernel<<<BATCH * C_IN, 256, 0, stream>>>(x, bn_scale, bn_shift, s);
  conv_wmma_kernel<<<dim3(L_LEN / 64, C_OUT / 128, BATCH), 256, 0, stream>>>(
      wh, s, (float*)d_out);
}
